// LSTM_forloop_53601191854439
// MI455X (gfx1250) — matrix operand order repaired and checked by reading, not compiled, not run on hardware
//
#include <hip/hip_runtime.h>
#include <hip/hip_bf16.h>
#include <math.h>

// LSTM: T=512, B=64, IN=512, H=1024.  bf16 WMMA (16x16x32), fp32 accum.
// Phase 1: convert to bf16, pregates = xs @ W_ih^T + (b_ih+b_hh) as one big WMMA GEMM.
// Phase 2: persistent recurrence: 64 blocks (one per 16-wide hidden tile jt) x 4 waves
//          (one per batch tile mt). Each block stages its W_hh column slice per
//          k-chunk into double-buffered LDS via global_load_async_to_lds_b128
//          (ASYNCcnt), all 4 waves consume identical B fragments from LDS.
//          Cell state c lives in VGPRs across all 512 steps; one grid-wide
//          atomic barrier per timestep.

typedef __bf16 bf16;
typedef __attribute__((ext_vector_type(16))) __bf16 v16bf;
typedef __attribute__((ext_vector_type(8)))  __bf16 v8bf;
typedef __attribute__((ext_vector_type(8)))  float  v8f;

#define T_  512
#define B_  64
#define IN_ 512
#define H_  1024
#define G4_ 4096   // 4*H
#define KC_ 32     // K-chunks per step (H/32)

__device__ inline float sigm(float x)      { return 1.0f / (1.0f + __expf(-x)); }
__device__ inline float tanh_fast(float x) { float e = __expf(-2.0f * x); return (1.0f - e) / (1.0f + e); }

// ISA 10.2: flat shared-pointer low 32 bits == LDS byte offset
__device__ inline unsigned lds_off(const void* p) { return (unsigned)(size_t)p; }

// A-matrix 16x32 bf16 fragment (contiguous k order, as in loadB): lane L holds row (L%16);
// VGPR0-3 <- K[k + (L/16)*16 .. +8), VGPR4-7 <- K[k+8+(L/16)*16 .. +8)
__device__ inline v16bf loadA(const bf16* __restrict__ base, int row, int ldk, int k, int hi) {
    const v8bf* p0 = (const v8bf*)(base + (size_t)row * ldk + k + hi * 16);
    const v8bf* p1 = (const v8bf*)(base + (size_t)row * ldk + k + 8 + hi * 16);
    v8bf a0 = *p0, a1 = *p1;
    return __builtin_shufflevector(a0, a1, 0,1,2,3,4,5,6,7,8,9,10,11,12,13,14,15);
}

// B-matrix 32x16 bf16 fragment from global: lane L holds column (L%16) = weight
// row `wrow`, 16 contiguous K values starting at k + (L/16)*16.
__device__ inline v16bf loadB(const bf16* __restrict__ base, int wrow, int ldk, int k, int hi) {
    const v16bf* p = (const v16bf*)(base + (size_t)wrow * ldk + k + hi * 16);
    return *p;
}

// async DMA: global (16B per lane) -> LDS, tracked with ASYNCcnt
__device__ inline void async_g2l_b128(unsigned ldsoff, unsigned long long gaddr) {
    asm volatile("global_load_async_to_lds_b128 %0, %1, off"
                 :: "v"(ldsoff), "v"(gaddr) : "memory");
}
__device__ inline void wait_async0() {
    asm volatile("s_wait_asynccnt 0x0" ::: "memory");
}

// ---------------- utility kernels ----------------
__global__ void f2bf_kernel(const float* __restrict__ s, bf16* __restrict__ d, int n) {
    int i = blockIdx.x * blockDim.x + threadIdx.x;
    if (i < n) d[i] = (bf16)s[i];
}
__global__ void bias_combine_kernel(const float* __restrict__ a, const float* __restrict__ b,
                                    float* __restrict__ d, int n) {
    int i = blockIdx.x * blockDim.x + threadIdx.x;
    if (i < n) d[i] = a[i] + b[i];
}
__global__ void reset_cnt_kernel(unsigned* c) {
    if (threadIdx.x == 0 && blockIdx.x == 0) *c = 0u;
}

// ---------------- phase 1: pregate GEMM ----------------
// preG[T*B, 4H] = xs_bf16[T*B, IN] @ W_ih_bf16[4H, IN]^T + bias  (one tile per wave)
__global__ __launch_bounds__(256) void pregate_gemm_kernel(const bf16* __restrict__ xsb,
                                                           const bf16* __restrict__ wih,
                                                           const float* __restrict__ bias,
                                                           float* __restrict__ preG) {
    const int lane = threadIdx.x & 31;
    const int ln = lane & 15, hi = lane >> 4;
    const int wave = blockIdx.x * (blockDim.x >> 5) + (threadIdx.x >> 5);
    const int tilesN = G4_ / 16;
    const int mt = wave / tilesN;      // 0..2047
    const int nt = wave % tilesN;      // 0..255

    const int arow = mt * 16 + ln;
    const int brow = nt * 16 + ln;

    v8f acc;
    const float bv = bias[nt * 16 + ln];
    #pragma unroll
    for (int v = 0; v < 8; v++) acc[v] = bv;

    for (int k = 0; k < IN_; k += 32) {
        v16bf a = loadA(xsb, arow, IN_, k, hi);
        v16bf b = loadB(wih, brow, IN_, k, hi);
        acc = __builtin_amdgcn_wmma_f32_16x16x32_bf16(false, a, false, b, (short)0, acc, false, false);
    }

    float* dst = preG + ((size_t)mt * 16 + 8 * hi) * G4_ + nt * 16 + ln;
    #pragma unroll
    for (int v = 0; v < 8; v++) dst[(size_t)v * G4_] = acc[v];
}

// ---------------- phase 2: persistent recurrence ----------------
__global__ __launch_bounds__(128) void lstm_recur_kernel(const float* __restrict__ preG,
                                                         const bf16* __restrict__ whh,
                                                         bf16* __restrict__ h0b,
                                                         bf16* __restrict__ h1b,
                                                         const float* __restrict__ c0,
                                                         float* __restrict__ out,
                                                         unsigned* __restrict__ cnt) {
    const int lane = threadIdx.x & 31;
    const int ln = lane & 15, hi = lane >> 4;
    const int wv = threadIdx.x >> 5;   // wave-in-block = batch tile mt (0..3)
    const int jt = blockIdx.x;         // hidden tile (0..63)
    const int m0 = wv * 16, j0 = jt * 16;

    // double-buffered B chunk: [buf][gate*32+lane] -> 16 bf16 (32B), 4KB per buffer
    __shared__ __align__(16) bf16 smem[2][2048];

    // ---- producer mapping: 256 16-byte segments per chunk, 2 per thread ----
    // segment s: gate g = s>>6, lane l = (s>>1)&31, half hh = s&1
    unsigned long long gsrc[2];   // global byte address at k=0
    unsigned           ldst[2];   // byte offset within an LDS buffer
    #pragma unroll
    for (int s2 = 0; s2 < 2; s2++) {
        unsigned s  = (unsigned)threadIdx.x + (unsigned)s2 * 128u;
        unsigned g  = s >> 6;
        unsigned l  = (s >> 1) & 31u;
        unsigned hh = s & 1u;
        unsigned row  = g * H_ + (unsigned)j0 + (l & 15u);      // W_hh row (= out col)
        unsigned koff = ((l >> 4) << 4) + (hh << 3);            // element offset in chunk
        gsrc[s2] = (unsigned long long)(whh + (size_t)row * H_ + koff);
        ldst[s2] = s * 16u;
    }
    const unsigned lbase[2] = { lds_off(&smem[0][0]), lds_off(&smem[1][0]) };

    // persistent cell state tile (C/D layout: m = 8*hi + v, n = ln)
    v8f c;
    {
        const float* cp = c0 + (size_t)(m0 + 8 * hi) * H_ + j0 + ln;
        #pragma unroll
        for (int v = 0; v < 8; v++) c[v] = cp[(size_t)v * H_];
    }

    bf16* hbuf[2] = { h0b, h1b };
    const unsigned nb = gridDim.x;

    for (int t = 0; t < T_; t++) {
        const bf16* hcur = hbuf[t & 1];
        bf16*       hnxt = hbuf[(t + 1) & 1];
        const float* pg = preG + (size_t)t * B_ * G4_;

        // kick off DMA of chunk 0 (buf 0). Safe WAR: buf0 last read at chunk 30
        // of step t-1, before that step's final barriers.
        #pragma unroll
        for (int s2 = 0; s2 < 2; s2++)
            async_g2l_b128(lbase[0] + ldst[s2], gsrc[s2]);

        if (t + 1 < T_)  // prefetch next step's pregate tile into cache
            __builtin_prefetch(preG + (size_t)(t + 1) * B_ * G4_ +
                               (size_t)(m0 + 8 * hi) * G4_ + j0 + ln, 0, 1);

        // seed accumulators with pregates (bias folded in)
        v8f acc[4];
        #pragma unroll
        for (int g = 0; g < 4; g++) {
            const float* p = pg + (size_t)(m0 + 8 * hi) * G4_ + g * H_ + j0 + ln;
            #pragma unroll
            for (int v = 0; v < 8; v++) acc[g][v] = p[(size_t)v * G4_];
        }

        // gates += h @ W_hh^T : B fragments via LDS (shared by all 4 waves),
        // A fragments per wave from global (L2-resident h).
        for (int kc = 0; kc < KC_; kc++) {
            wait_async0();        // our chunk-kc segments have landed in LDS
            __syncthreads();      // all threads' segments visible; prior reads done
            if (kc + 1 < KC_) {   // overlap DMA of next chunk with WMMA of this one
                const int nb2 = (kc + 1) & 1;
                const unsigned long long koffb = (unsigned long long)(kc + 1) * 32ull * sizeof(bf16);
                #pragma unroll
                for (int s2 = 0; s2 < 2; s2++)
                    async_g2l_b128(lbase[nb2] + ldst[s2], gsrc[s2] + koffb);
            }
            v16bf a = loadA(hcur, m0 + ln, H_, kc * 32, hi);
            const bf16* bb = &smem[kc & 1][0];
            #pragma unroll
            for (int g = 0; g < 4; g++) {
                v16bf b = *(const v16bf*)(bb + (g * 32 + lane) * 16);
                acc[g] = __builtin_amdgcn_wmma_f32_16x16x32_bf16(false, a, false, b, (short)0, acc[g], false, false);
            }
        }

        // pointwise LSTM cell (gate order i,f,g,o); write h fp32 + bf16
        float* op = out + ((size_t)t * B_ + m0 + 8 * hi) * H_ + j0 + ln;
        bf16*  hp = hnxt + (size_t)(m0 + 8 * hi) * H_ + j0 + ln;
        #pragma unroll
        for (int v = 0; v < 8; v++) {
            float ig = sigm(acc[0][v]);
            float fg = sigm(acc[1][v]);
            float gg = tanh_fast(acc[2][v]);
            float og = sigm(acc[3][v]);
            float cn = fg * c[v] + ig * gg;
            c[v] = cn;
            float h = og * tanh_fast(cn);
            op[(size_t)v * H_] = h;
            hp[(size_t)v * H_] = (bf16)h;
        }

        // grid-wide barrier (monotonic counter; reset each launch)
        __threadfence();
        __syncthreads();
        if (threadIdx.x == 0) {
            atomicAdd(cnt, 1u);
            const unsigned target = nb * (unsigned)(t + 1);
            while (__hip_atomic_load(cnt, __ATOMIC_ACQUIRE, __HIP_MEMORY_SCOPE_AGENT) < target) {
                __builtin_amdgcn_s_sleep(2);
            }
        }
        __syncthreads();
    }
}

extern "C" void kernel_launch(void* const* d_in, const int* in_sizes, int n_in,
                              void* d_out, int out_size, void* d_ws, size_t ws_size,
                              hipStream_t stream) {
    const float* xs  = (const float*)d_in[0];   // [T,B,IN]
    const float* h0  = (const float*)d_in[1];   // [B,H]
    const float* c0  = (const float*)d_in[2];   // [B,H]
    const float* Wih = (const float*)d_in[3];   // [4H,IN]
    const float* Whh = (const float*)d_in[4];   // [4H,H]
    const float* bih = (const float*)d_in[5];   // [4H]
    const float* bhh = (const float*)d_in[6];   // [4H]
    float* out = (float*)d_out;                 // [T,B,H]

    char* ws = (char*)d_ws;
    size_t off = 0;
    auto carve = [&](size_t bytes) -> char* {
        char* p = ws + off;
        off += (bytes + 255) & ~(size_t)255;
        return p;
    };
    bf16*     xsb   = (bf16*)    carve((size_t)T_ * B_ * IN_ * sizeof(bf16));  // 32 MB
    bf16*     wihb  = (bf16*)    carve((size_t)G4_ * IN_ * sizeof(bf16));      // 4 MB
    bf16*     whhb  = (bf16*)    carve((size_t)G4_ * H_ * sizeof(bf16));       // 8 MB
    float*    bias  = (float*)   carve((size_t)G4_ * sizeof(float));
    bf16*     hbuf0 = (bf16*)    carve((size_t)B_ * H_ * sizeof(bf16));
    bf16*     hbuf1 = (bf16*)    carve((size_t)B_ * H_ * sizeof(bf16));
    unsigned* cnt   = (unsigned*)carve(sizeof(unsigned));
    float*    preG  = (float*)   carve((size_t)T_ * B_ * G4_ * sizeof(float)); // 512 MB

    const int thr = 256;
    int n;
    n = T_ * B_ * IN_; f2bf_kernel<<<(n + thr - 1) / thr, thr, 0, stream>>>(xs,  xsb,  n);
    n = G4_ * IN_;     f2bf_kernel<<<(n + thr - 1) / thr, thr, 0, stream>>>(Wih, wihb, n);
    n = G4_ * H_;      f2bf_kernel<<<(n + thr - 1) / thr, thr, 0, stream>>>(Whh, whhb, n);
    n = B_ * H_;       f2bf_kernel<<<(n + thr - 1) / thr, thr, 0, stream>>>(h0,  hbuf0, n);
    n = G4_;           bias_combine_kernel<<<(n + thr - 1) / thr, thr, 0, stream>>>(bih, bhh, bias, n);
    reset_cnt_kernel<<<1, 32, 0, stream>>>(cnt);

    // Phase 1: 2048 x 256 tiles, one tile per wave, 8 waves per block
    {
        const int totalTiles = (T_ * B_ / 16) * (G4_ / 16);
        pregate_gemm_kernel<<<totalTiles / 8, 256, 0, stream>>>(xsb, wihb, bias, preG);
    }

    // Phase 2: persistent recurrence, 64 blocks (jt) x 128 threads (4 mt waves)
    lstm_recur_kernel<<<64, 128, 0, stream>>>(preG, whhb, hbuf0, hbuf1, c0, out, cnt);
}
